// RPN_82781199663693
// MI455X (gfx1250) — compile-verified, matrix operand order checked by reading, not run on hardware
//
#include <hip/hip_runtime.h>
#include <hip/hip_bf16.h>

typedef __bf16 bf16_t;
typedef __attribute__((ext_vector_type(16))) __bf16 v16bf;
typedef __attribute__((ext_vector_type(8)))  __bf16 v8bf;
typedef __attribute__((ext_vector_type(8)))  float  v8f;

#define C_IN    2048
#define C_MID   1024
#define KB      32              // K step for wmma_f32_16x16x32_bf16
#define LDA     40              // padded LDS row stride (bf16); rows stay 16B-aligned
#define NSTEP   576             // 9 taps * (2048/32)
#define BTW_BYTES  (9ull * 1024ull * 2048ull * 2ull)     // 37,748,736
#define XPAD_BYTES (32ull * 256ull * 2048ull * 2ull)     // 33,554,432

static __device__ __forceinline__ bf16_t f2bf(float f) {
    unsigned u = __builtin_bit_cast(unsigned, f);
    u += 0x7fffu + ((u >> 16) & 1u);            // round-to-nearest-even
    unsigned short h = (unsigned short)(u >> 16);
    return __builtin_bit_cast(bf16_t, h);
}

// ---------------------------------------------------------------------------
// Kernel 1: w1 [1024,2048,3,3] f32 (OIHW) -> Btw[r][n][c] bf16.
//   One thread per (n,c): 9 contiguous f32 reads, 9 coalesced bf16 writes.
// ---------------------------------------------------------------------------
__global__ __launch_bounds__(256)
void prep_weights(const float* __restrict__ w1, bf16_t* __restrict__ Btw) {
    int o = blockIdx.x * 256 + threadIdx.x;     // (n,c) pair, < 1024*2048
    int n = o >> 11;
    int c = o & 2047;
    const float* src = w1 + (size_t)o * 9;
    #pragma unroll
    for (int r = 0; r < 9; ++r)
        Btw[((size_t)r * 1024 + n) * 2048 + c] = f2bf(src[r]);
}

// ---------------------------------------------------------------------------
// Kernel 2: x NCHW f32 -> halo-padded NHWC bf16: xpad[n][ih+1][iw+1][c],
//   16x16 spatial with zero border. LDS-tiled transpose, both sides coalesced.
// ---------------------------------------------------------------------------
__global__ __launch_bounds__(256)
void prep_x(const float* __restrict__ x, bf16_t* __restrict__ xpad) {
    __shared__ bf16_t sT[64 * 200];             // [cc][s], padded rows
    const int t  = threadIdx.x;
    const int n  = blockIdx.x;                  // image
    const int c0 = blockIdx.y * 64;             // channel tile

    const float* src = x + ((size_t)n * C_IN + c0) * 196;
    #pragma unroll
    for (int i = 0; i < 49; ++i) {              // 49*256 = 64*196 exactly
        int idx = t + 256 * i;
        int cc  = idx / 196;
        int s   = idx - cc * 196;
        sT[cc * 200 + s] = f2bf(src[(size_t)cc * 196 + s]);
    }
    __syncthreads();

    const bf16_t zero = __builtin_bit_cast(bf16_t, (unsigned short)0);
    bf16_t* dst = xpad + (size_t)n * (256 * 2048) + c0;
    #pragma unroll 4
    for (int i = 0; i < 64; ++i) {              // 64*256 = 256*64 exactly
        int o   = t + 256 * i;
        int cc  = o & 63;                       // consecutive threads -> consecutive c
        int sp  = o >> 6;                       // padded spatial 0..255
        int ih  = (sp >> 4) - 1;
        int iw  = (sp & 15) - 1;
        bool ok = ((unsigned)ih < 14u) && ((unsigned)iw < 14u);
        bf16_t v = ok ? sT[cc * 200 + ih * 14 + iw] : zero;
        dst[(size_t)sp * 2048 + cc] = v;
    }
}

// ---------------------------------------------------------------------------
// Kernel 3: implicit-GEMM 3x3 conv via bf16 WMMA, double-buffered LDS,
//   branch-free vectorized staging with incremental global pointers,
//   fused bias+BN+ReLU epilogue.
//   Block: 256 thr = 8 waves; tile 128x128; wave grid 2(m) x 4(n);
//   wave computes 4x2 tiles of 16x16 -> 8 x v_wmma_f32_16x16x32_bf16 / step.
// ---------------------------------------------------------------------------
__global__ __launch_bounds__(256)
void conv3x3_wmma(const bf16_t* __restrict__ xpad,
                  const bf16_t* __restrict__ Btw,
                  const float* __restrict__ b1,
                  const float* __restrict__ g1,  const float* __restrict__ be1,
                  const float* __restrict__ mu1, const float* __restrict__ va1,
                  float* __restrict__ hmid)
{
    __shared__ bf16_t sA[2][128 * LDA];
    __shared__ bf16_t sB[2][128 * LDA];
    __shared__ float  sInv[128];
    __shared__ float  sAdd[128];

    const int tid   = threadIdx.x;
    const int lane  = tid & 31;
    const int wid   = tid >> 5;
    const int wm    = wid & 1;
    const int wn    = wid >> 1;
    const int row   = lane & 15;
    const int halfk = lane >> 4;

    const int mBlock = blockIdx.x * 128;
    const int nBlock = blockIdx.y * 128;

    if (tid < 128) {
        int n = nBlock + tid;
        float inv = g1[n] * rsqrtf(va1[n] + 1e-5f);
        sInv[tid] = inv;
        sAdd[tid] = be1[n] - mu1[n] * inv + b1[n] * inv;
    }

    // ---- staging geometry (constant across all steps) ----
    const int mloc  = tid & 127;                 // tile row this thread stages
    const int chunk = tid >> 7;                  // which 16-elem c chunk
    const int gm    = mBlock + mloc;
    const int nimg  = gm / 196;
    const int pix   = gm - nimg * 196;
    const int oh    = pix / 14;
    const int ow    = pix - oh * 14;

    // incremental global pointers: +32 elems/step; tap boundary corrections.
    const bf16_t* pa = xpad + ((size_t)nimg * 256 + oh * 16 + ow) * 2048 + chunk * 16;
    const bf16_t* pb = Btw + ((size_t)(nBlock + mloc)) * 2048 + chunk * 16;
    int t64 = 0;                                 // steps within current tap
    int t3  = 0;                                 // taps within current kernel row

    const int sOff = mloc * LDA + chunk * 16;    // LDS staging offset (A and B)

    v8f acc[4][2];
    #pragma unroll
    for (int a = 0; a < 4; ++a)
        #pragma unroll
        for (int b = 0; b < 2; ++b)
            #pragma unroll
            for (int v = 0; v < 8; ++v)
                acc[a][b][v] = 0.f;

    v8bf ra0, ra1, rb0, rb1;                     // staged registers

    auto load_next = [&]() {
        ra0 = *(const v8bf*)pa;
        ra1 = *(const v8bf*)(pa + 8);
        rb0 = *(const v8bf*)pb;
        rb1 = *(const v8bf*)(pb + 8);
        pa += KB;                                // next c chunk (64 B)
        pb += KB;
        if (++t64 == 64) {                       // tap boundary (uniform branch)
            t64 = 0;
            pb += (1024 * 2048 - 2048);          // next r plane of Btw
            if (++t3 == 3) {                     // kernel-row wrap: +13 pixels to next padded row
                t3 = 0;
                pa += 13 * 2048;
            }
        }
    };
    auto store_step = [&](int buf) {
        *(v8bf*)&sA[buf][sOff]     = ra0;
        *(v8bf*)&sA[buf][sOff + 8] = ra1;
        *(v8bf*)&sB[buf][sOff]     = rb0;
        *(v8bf*)&sB[buf][sOff + 8] = rb1;
    };

    load_next();
    store_step(0);

    #pragma unroll 2
    for (int step = 0; step < NSTEP; ++step) {
        const int  cur  = step & 1;
        const bool more = (step + 1) < NSTEP;
        if (more) load_next();                   // global loads overlap compute
        __syncthreads();                         // buf[cur] staged; prev reads done
        if (more) store_step(1 - cur);           // fill idle buffer

        v16bf afrag[4];
        #pragma unroll
        for (int tm = 0; tm < 4; ++tm) {
            const bf16_t* qa = &sA[cur][(wm * 64 + tm * 16 + row) * LDA + halfk * 8];
            v8bf lo = *(const v8bf*)qa;          // k = halfk*8 .. +7
            v8bf hi = *(const v8bf*)(qa + 16);   // k = 16+halfk*8 .. +7
            afrag[tm] = __builtin_shufflevector(lo, hi,
                0,1,2,3,4,5,6,7,8,9,10,11,12,13,14,15);
        }
        v16bf bfrag[2];
        #pragma unroll
        for (int tn = 0; tn < 2; ++tn) {
            const bf16_t* qb = &sB[cur][(wn * 32 + tn * 16 + row) * LDA + halfk * 8];
            v8bf lo = *(const v8bf*)qb;
            v8bf hi = *(const v8bf*)(qb + 16);
            bfrag[tn] = __builtin_shufflevector(lo, hi,
                0,1,2,3,4,5,6,7,8,9,10,11,12,13,14,15);
        }
        #pragma unroll
        for (int tm = 0; tm < 4; ++tm)
            #pragma unroll
            for (int tn = 0; tn < 2; ++tn)
                acc[tm][tn] = __builtin_amdgcn_wmma_f32_16x16x32_bf16(
                    false, afrag[tm], false, bfrag[tn],
                    (short)0, acc[tm][tn], false, false);
    }

    // ---- fused epilogue: bias + BN + ReLU ; write h_mid[m, 1024] ----
    #pragma unroll
    for (int tm = 0; tm < 4; ++tm) {
        #pragma unroll
        for (int tn = 0; tn < 2; ++tn) {
            const int   nl  = wn * 32 + tn * 16 + row;   // C/D: N = lane%16
            const int   gn  = nBlock + nl;
            const float inv = sInv[nl];
            const float add = sAdd[nl];
            #pragma unroll
            for (int v = 0; v < 8; ++v) {                // C/D: M = v + 8*(lane/16)
                int gmr = mBlock + wm * 64 + tm * 16 + v + 8 * halfk;
                float h = fmaxf(acc[tm][tn][v] * inv + add, 0.f);
                hmid[(size_t)gmr * C_MID + gn] = h;
            }
        }
    }
}

// ---------------------------------------------------------------------------
// Kernel 4: 1x1 heads + softmax + NHWC anchor layout (wave-per-pixel).
// ---------------------------------------------------------------------------
__global__ __launch_bounds__(256)
void rpn_head(const float* __restrict__ hmid,
              const float* __restrict__ w_sel, const float* __restrict__ b_sel,
              const float* __restrict__ w_box, const float* __restrict__ b_box,
              const float* __restrict__ gb,  const float* __restrict__ beb,
              const float* __restrict__ mub, const float* __restrict__ vab,
              float* __restrict__ out)
{
    __shared__ float sdot[8][56];
    const int lane = threadIdx.x & 31;
    const int w    = threadIdx.x >> 5;
    const int p    = blockIdx.x * 8 + w;          // pixel index, 0..6271

    const float* hp = hmid + (size_t)p * C_MID;
    float hreg[32];
    #pragma unroll
    for (int i = 0; i < 32; ++i) hreg[i] = hp[lane + 32 * i];

    for (int j = 0; j < 54; ++j) {
        const float* wr = (j < 18) ? (w_sel + j * C_MID)
                                   : (w_box + (j - 18) * C_MID);
        float acc = 0.f;
        #pragma unroll
        for (int i = 0; i < 32; ++i) acc += hreg[i] * wr[lane + 32 * i];
        #pragma unroll
        for (int off = 16; off > 0; off >>= 1) acc += __shfl_xor(acc, off, 32);
        if (lane == 0) sdot[w][j] = acc;
    }
    __syncthreads();

    const int nimg = p / 196;
    const int pix  = p - nimg * 196;
    const size_t score_base = (size_t)nimg * (1764 * 2) + (size_t)pix * 18;
    const size_t box_base   = 112896u + (size_t)nimg * (1764 * 4) + (size_t)pix * 36;

    if (lane < 9) {
        float s0 = fmaxf(sdot[w][2 * lane]     + b_sel[2 * lane],     0.f);
        float s1 = fmaxf(sdot[w][2 * lane + 1] + b_sel[2 * lane + 1], 0.f);
        float m  = fmaxf(s0, s1);
        float e0 = __expf(s0 - m), e1 = __expf(s1 - m);
        float rn = 1.f / (e0 + e1);
        out[score_base + 2 * lane]     = e0 * rn;
        out[score_base + 2 * lane + 1] = e1 * rn;
    }
    for (int j = lane; j < 36; j += 32) {
        float v   = sdot[w][18 + j] + b_box[j];
        float inv = gb[j] * rsqrtf(vab[j] + 1e-5f);
        v = v * inv + (beb[j] - mub[j] * inv);
        out[box_base + j] = fmaxf(v, 0.f) * 419.0f;
    }
}

// ---------------------------------------------------------------------------
extern "C" void kernel_launch(void* const* d_in, const int* in_sizes, int n_in,
                              void* d_out, int out_size, void* d_ws, size_t ws_size,
                              hipStream_t stream) {
    const float* x    = (const float*)d_in[0];
    const float* w1   = (const float*)d_in[1];
    const float* b1   = (const float*)d_in[2];
    const float* g1   = (const float*)d_in[3];
    const float* be1  = (const float*)d_in[4];
    const float* mu1  = (const float*)d_in[5];
    const float* va1  = (const float*)d_in[6];
    const float* wsel = (const float*)d_in[7];
    const float* bsel = (const float*)d_in[8];
    const float* wbox = (const float*)d_in[9];
    const float* bbox = (const float*)d_in[10];
    const float* gb   = (const float*)d_in[11];
    const float* beb  = (const float*)d_in[12];
    const float* mub  = (const float*)d_in[13];
    const float* vab  = (const float*)d_in[14];
    float* out = (float*)d_out;

    char*   ws   = (char*)d_ws;
    bf16_t* Btw  = (bf16_t*)ws;                              // 37.75 MB
    bf16_t* xpad = (bf16_t*)(ws + BTW_BYTES);                // 33.55 MB
    float*  hmid = (float*)(ws + BTW_BYTES + XPAD_BYTES);    // 25.69 MB

    // 1) weight transform: OIHW f32 -> [r][n][c] bf16
    prep_weights<<<(1024 * 2048) / 256, 256, 0, stream>>>(w1, Btw);
    // 2) input transform: NCHW f32 -> halo-padded NHWC bf16
    prep_x<<<dim3(32, 32), 256, 0, stream>>>(x, xpad);
    // 3) implicit-GEMM 3x3 conv via bf16 WMMA (M=6272, N=1024, K=18432)
    conv3x3_wmma<<<dim3(49, 8), 256, 0, stream>>>(xpad, Btw, b1, g1, be1, mu1, va1, hmid);
    // 4) heads + softmax + anchor layout
    rpn_head<<<784, 256, 0, stream>>>(hmid, wsel, bsel, wbox, bbox,
                                      gb, beb, mub, vab, out);
}
